// Model_13984413516255
// MI455X (gfx1250) — matrix operand order repaired and checked by reading, not compiled, not run on hardware
//
#include <hip/hip_runtime.h>
#include <hip/hip_bf16.h>
#include <math.h>

// ---------------------------------------------------------------------------
// MI455X (gfx1250) implementation.
//  - rfft/irfft expressed as dense DFT GEMMs -> v_wmma_f32_16x16x32_f16
//  - fft_convolve collapsed to a 128-tap direct convolution (upsample holes)
//  - dilated blocks: 1 workgroup per batch, whole state in LDS (wave32, WGP)
// ---------------------------------------------------------------------------

#define NSAMP   32768
#define STEPSZ  256
#define WIN     2048
#define NFR     128          // N_FRAMES
#define NCO     1025         // N_COEFFS
#define NKEEP   1024         // coeffs kept in spec
#define NB      32           // batch
#define CHN     32           // hidden channels
#define KPAD    1056         // 1025 padded to multiple of 32

typedef __attribute__((ext_vector_type(16))) _Float16 v16h;
typedef __attribute__((ext_vector_type(8)))  _Float16 v8h;
typedef __attribute__((ext_vector_type(8)))  float    v8f;

// ---- workspace layout (bytes, 256-aligned) --------------------------------
#define OFF_HANN    ((size_t)0)                        //  2048 f32
#define OFF_B2      ((size_t)8192)                     //  2048x2048 f16 (cos|sin, col-major)
#define OFF_ICOS    ((size_t)8396800)                  //  2048x1056 f16 (irfft table, col-major)
#define OFF_FRAMES  ((size_t)12722176)                 //  4096x2048 f16  (later reused for MAGS)
#define OFF_C1      ((size_t)29499392)                 //  4096x2048 f32  (later reused for G)
#define OFF_SPEC    ((size_t)63053824)                 //  32x1024x128 f32
#define OFF_H       ((size_t)79831040)                 //  32x32x128 f32
#define OFF_ACC     ((size_t)80355328)                 //  32x32x128 f32
#define OFF_XV      ((size_t)80879616)                 //  32x32x128 f32
#define OFF_RES     ((size_t)81403904)                 //  32x32768 f32
#define OFF_MAGS    OFF_FRAMES                         //  4096x1056 f16 (reuse)
#define OFF_G       OFF_C1                             //  4096x2048 f32 (reuse)

#define PI2_OVER_WIN 0.0030679615757712823f            // 2*pi/2048

// ---------------------------------------------------------------------------
__global__ void k_hann(float* __restrict__ hann) {
  int n = blockIdx.x * blockDim.x + threadIdx.x;
  if (n < WIN) hann[n] = 0.5f - 0.5f * cosf(PI2_OVER_WIN * (float)n);
}

// B2[col][n], col-major over K=n: col<1024 -> cos(2*pi*k*n/2048), else sin.
__global__ void k_b2(_Float16* __restrict__ b2) {
  int idx = blockIdx.x * blockDim.x + threadIdx.x;
  if (idx >= WIN * WIN) return;
  int kk = idx >> 11, n = idx & 2047;
  int k  = (kk < NKEEP) ? kk : (kk - NKEEP);
  int m  = (k * n) & 2047;                 // exact periodic reduction
  float ang = PI2_OVER_WIN * (float)m;
  float v = (kk < NKEEP) ? cosf(ang) : sinf(ang);
  b2[idx] = (_Float16)v;
}

// icos[n][k]: irfft basis with backward-norm scaling baked in.
__global__ void k_icos(_Float16* __restrict__ icos) {
  int idx = blockIdx.x * blockDim.x + threadIdx.x;
  if (idx >= WIN * KPAD) return;
  int n = idx / KPAD, k = idx - n * KPAD;
  float v;
  const float s = 1.0f / (float)WIN;
  if (k == 0)          v = s;
  else if (k < NKEEP)  { int m = (k * n) & 2047; v = 2.0f * s * cosf(PI2_OVER_WIN * (float)m); }
  else if (k == NKEEP) v = s * ((n & 1) ? -1.0f : 1.0f);
  else                 v = 0.0f;
  icos[idx] = (_Float16)v;
}

// F[(b,f)][n] = hann[n] * audio[b, 256 f + n]  (zero beyond 32768)
__global__ void k_frames(const float* __restrict__ audio, const float* __restrict__ hann,
                         _Float16* __restrict__ F) {
  int idx = blockIdx.x * blockDim.x + threadIdx.x;
  if (idx >= NB * NFR * WIN) return;
  int n = idx & 2047, row = idx >> 11;
  int b = row >> 7, f = row & 127;
  int s = f * STEPSZ + n;
  float v = (s < NSAMP) ? audio[(size_t)b * NSAMP + s] * hann[n] : 0.0f;
  F[idx] = (_Float16)v;
}

// ---------------------------------------------------------------------------
// Generic f16 WMMA GEMM:  C[MxN] = A[MxK, row-major] * B[KxN, col-major (Bc[n*K+k])]
// One wave per 16x16 tile; K-loop in steps of 32 with v_wmma_f32_16x16x32_f16.
// Fragment layouts per CDNA5 ISA 7.12.2.
__global__ void __launch_bounds__(256)
k_gemm(const _Float16* __restrict__ A, const _Float16* __restrict__ Bc,
       float* __restrict__ C, int M, int N, int K) {
  int wave = (int)((blockIdx.x * blockDim.x + threadIdx.x) >> 5);
  int lane = threadIdx.x & 31;
  int ntiles = N >> 4;
  int tm = wave / ntiles;
  int tn = wave - tm * ntiles;
  if (tm >= (M >> 4)) return;                 // wave-uniform -> EXEC stays all-1
  int l15 = lane & 15;
  int hi  = lane >> 4;
  const _Float16* arow = A  + (size_t)(tm * 16 + l15) * K + hi * 8;   // k runs: [hi*8, +8) and [hi*8+16, +8)
  const _Float16* bcol = Bc + (size_t)(tn * 16 + l15) * K + hi * 8;   // k runs: [hi*8, +8) and [hi*8+16, +8)
  v8f acc = {};
  for (int k0 = 0; k0 < K; k0 += 32) {
    v8h a0 = *(const v8h*)(arow + k0);
    v8h a1 = *(const v8h*)(arow + k0 + 16);
    v8h b0 = *(const v8h*)(bcol + k0);
    v8h b1 = *(const v8h*)(bcol + k0 + 16);
    __builtin_prefetch(arow + k0 + 128, 0, 1);   // global_prefetch_b8
    __builtin_prefetch(bcol + k0 + 128, 0, 1);
    v16h a, b;
#pragma unroll
    for (int j = 0; j < 8; ++j) { a[j] = a0[j]; a[j + 8] = a1[j]; b[j] = b0[j]; b[j + 8] = b1[j]; }
    acc = __builtin_amdgcn_wmma_f32_16x16x32_f16(false, a, false, b, (short)0, acc, false, false);
  }
  // D layout: VGPR r -> M = r + hi*8 ; N = l15
  float* cp = C + (size_t)(tm * 16 + hi * 8) * N + tn * 16 + l15;
#pragma unroll
  for (int r = 0; r < 8; ++r) cp[(size_t)r * N] = acc[r];
}

// spec[b][k][f] = |rfft|/sqrt(2048) from C1[(b,f)][k (re) / k+1024 (im)]
__global__ void k_spec_mag(const float* __restrict__ C1, float* __restrict__ spec) {
  int idx = blockIdx.x * blockDim.x + threadIdx.x;
  if (idx >= NB * NKEEP * NFR) return;
  int f = idx & 127;
  int k = (idx >> 7) & 1023;
  int b = idx >> 17;
  size_t row = (size_t)(b * NFR + f) * WIN;
  float re = C1[row + k], im = C1[row + NKEEP + k];
  spec[idx] = sqrtf(re * re + im * im) * 0.0220970869120796f;  // 1/sqrt(2048)
}

// h[b][o][t] = b_in[o] + sum_i w_in[o][i] * spec[b][i][t]
__global__ void k_proj_in(const float* __restrict__ spec, const float* __restrict__ w,
                          const float* __restrict__ bias, float* __restrict__ h) {
  int idx = blockIdx.x * blockDim.x + threadIdx.x;
  if (idx >= NB * CHN * NFR) return;
  int t = idx & 127, o = (idx >> 7) & 31, b = idx >> 12;
  float s = bias[o];
  const float* sp = spec + (size_t)b * NKEEP * NFR + t;
  const float* wr = w + (size_t)o * NKEEP;
  for (int i = 0; i < NKEEP; ++i) s += wr[i] * sp[(size_t)i * NFR];
  h[idx] = s;
}

// 8 dilated anticausal blocks; one workgroup per batch; state in LDS.
__global__ void __launch_bounds__(256)
k_blocks(const float* __restrict__ hin, const float* __restrict__ bw,
         const float* __restrict__ bb, float* __restrict__ accOut) {
  __shared__ float sh_h[CHN][NFR];
  __shared__ float sh_y[CHN][NFR];
  __shared__ float sh_a[CHN][NFR];
  __shared__ float red[256];
  const int dil[8] = {1, 2, 4, 8, 16, 32, 64, 1};
  int b = blockIdx.x, tid = threadIdx.x;
  for (int e = tid; e < CHN * NFR; e += 256) {
    sh_h[e >> 7][e & 127] = hin[(size_t)b * CHN * NFR + e];
    sh_a[e >> 7][e & 127] = 0.0f;
  }
  __syncthreads();
  for (int blk = 0; blk < 8; ++blk) {
    int d = dil[blk];
    float lmax = 0.0f;
    for (int e = tid; e < CHN * NFR; e += 256) {
      int c = e >> 7, t = e & 127;
      float s = bb[blk * CHN + c];
      const float* wr = bw + (size_t)(blk * CHN + c) * CHN * 2;
      bool fut = (t + d) < NFR;
#pragma unroll 8
      for (int i = 0; i < CHN; ++i) {
        s += wr[2 * i] * sh_h[i][t];
        if (fut) s += wr[2 * i + 1] * sh_h[i][t + d];
      }
      float y = (s > 0.0f) ? s : 0.2f * s;   // leaky_relu 0.2
      y += sh_h[c][t];                       // residual
      sh_y[c][t] = y;
      lmax = fmaxf(lmax, fabsf(y));
    }
    __syncthreads();
    red[tid] = lmax;
    __syncthreads();
    for (int s2 = 128; s2 > 0; s2 >>= 1) {
      if (tid < s2) red[tid] = fmaxf(red[tid], red[tid + s2]);
      __syncthreads();
    }
    float inv = 1.0f / (red[0] + 1e-8f);
    __syncthreads();
    for (int e = tid; e < CHN * NFR; e += 256) {
      int c = e >> 7, t = e & 127;
      float v = sh_y[c][t] * inv;
      sh_h[c][t] = v;
      sh_a[c][t] += v;
    }
    __syncthreads();
  }
  for (int e = tid; e < CHN * NFR; e += 256)
    accOut[(size_t)b * CHN * NFR + e] = sh_a[e >> 7][e & 127];
}

// sparse[b][o][t] = relu(psb[o] + sum_i psw[o][i]*acc[b][i][t])   (o<256)
__global__ void k_proj_sparse(const float* __restrict__ acc, const float* __restrict__ w,
                              const float* __restrict__ bias, float* __restrict__ sparse) {
  int idx = blockIdx.x * blockDim.x + threadIdx.x;
  if (idx >= NB * 256 * NFR) return;
  int t = idx & 127, o = (idx >> 7) & 255, b = idx >> 15;
  float s = bias[o];
  const float* ar = acc + (size_t)b * CHN * NFR + t;
  const float* wr = w + (size_t)o * CHN;
#pragma unroll 8
  for (int i = 0; i < CHN; ++i) s += wr[i] * ar[(size_t)i * NFR];
  sparse[idx] = (s > 0.0f) ? s : 0.0f;
}

// xv[b][o][f] = pdb[o] + sum_i pdw[o][i]*sparse[b][i][f]   (i<256)
__global__ void k_proj_dense(const float* __restrict__ sparse, const float* __restrict__ w,
                             const float* __restrict__ bias, float* __restrict__ xv) {
  int idx = blockIdx.x * blockDim.x + threadIdx.x;
  if (idx >= NB * CHN * NFR) return;
  int f = idx & 127, o = (idx >> 7) & 31, b = idx >> 12;
  float s = bias[o];
  const float* sp = sparse + (size_t)b * 256 * NFR + f;
  const float* wr = w + (size_t)o * 256;
  for (int i = 0; i < 256; ++i) s += wr[i] * sp[(size_t)i * NFR];
  xv[idx] = s;
}

// mags[(c,f)][k] = clip(reso[c][k],0,0.9999)^(f+1)  (f16, K padded to 1056)
__global__ void k_mags(const float* __restrict__ reso, _Float16* __restrict__ mags) {
  int idx = blockIdx.x * blockDim.x + threadIdx.x;
  if (idx >= CHN * KPAD) return;
  int c = idx / KPAD, k = idx - c * KPAD;
  if (k >= NCO) {
    for (int f = 0; f < NFR; ++f) mags[(size_t)(c * NFR + f) * KPAD + k] = (_Float16)0.0f;
    return;
  }
  float base = reso[(size_t)c * NCO + k];
  base = fminf(fmaxf(base, 0.0f), 0.9999f);
  float p = 1.0f;
  for (int f = 0; f < NFR; ++f) {
    p *= base;
    mags[(size_t)(c * NFR + f) * KPAD + k] = (_Float16)p;
  }
}

// overlap-add (hop=1024): res[c][t] = sum of <=2 windowed frames from G
__global__ void k_ola(const float* __restrict__ G, const float* __restrict__ hann,
                      float* __restrict__ res) {
  int idx = blockIdx.x * blockDim.x + threadIdx.x;
  if (idx >= CHN * NSAMP) return;
  int c = idx >> 15, t = idx & (NSAMP - 1);
  int q = t >> 10, n0 = t & 1023;
  float a = G[(size_t)(c * NFR + q) * WIN + n0] * hann[n0];
  if (q > 0) {
    int n1 = n0 + 1024;
    a += G[(size_t)(c * NFR + q - 1) * WIN + n1] * hann[n1];
  }
  res[idx] = a;
}

// y[b][t] = sum_c sum_{f<=t/256} xv[b][c][f] * res[c][t-256f]  (128-tap sparse conv)
__global__ void __launch_bounds__(256)
k_conv(const float* __restrict__ xv, const float* __restrict__ res, float* __restrict__ y) {
  __shared__ float xs[CHN][NFR];
  int b = blockIdx.x >> 7;
  int tile = blockIdx.x & 127;          // == t >> 8 for all threads in block
  int tid = threadIdx.x;
  for (int e = tid; e < CHN * NFR; e += 256)
    xs[e >> 7][e & 127] = xv[(size_t)b * CHN * NFR + e];
  __syncthreads();
  int t = tile * 256 + tid;
  float s = 0.0f;
  for (int dq = 0; dq <= tile; ++dq) {
    int f = tile - dq;
    const float* rp = res + (size_t)dq * 256 + tid;   // + c*NSAMP below
#pragma unroll 8
    for (int c = 0; c < CHN; ++c) s += xs[c][f] * rp[(size_t)c * NSAMP];
  }
  y[(size_t)b * NSAMP + t] = s;
}

// ---------------------------------------------------------------------------
extern "C" void kernel_launch(void* const* d_in, const int* in_sizes, int n_in,
                              void* d_out, int out_size, void* d_ws, size_t ws_size,
                              hipStream_t stream) {
  const float* audio = (const float*)d_in[0];
  const float* w_in  = (const float*)d_in[1];
  const float* b_in  = (const float*)d_in[2];
  const float* bw    = (const float*)d_in[3];
  const float* bb    = (const float*)d_in[4];
  const float* psw   = (const float*)d_in[5];
  const float* psb   = (const float*)d_in[6];
  const float* pdw   = (const float*)d_in[7];
  const float* pdb   = (const float*)d_in[8];
  const float* reso  = (const float*)d_in[9];

  float* out_y      = (float*)d_out;                       // 32*1*32768
  float* out_sparse = (float*)d_out + (size_t)NB * NSAMP;  // 32*256*128

  char* ws = (char*)d_ws;
  float*     hann  = (float*)(ws + OFF_HANN);
  _Float16*  b2    = (_Float16*)(ws + OFF_B2);
  _Float16*  icos  = (_Float16*)(ws + OFF_ICOS);
  _Float16*  F     = (_Float16*)(ws + OFF_FRAMES);
  float*     C1    = (float*)(ws + OFF_C1);
  float*     spec  = (float*)(ws + OFF_SPEC);
  float*     h     = (float*)(ws + OFF_H);
  float*     acc   = (float*)(ws + OFF_ACC);
  float*     xv    = (float*)(ws + OFF_XV);
  float*     res   = (float*)(ws + OFF_RES);
  _Float16*  mags  = (_Float16*)(ws + OFF_MAGS);           // reuses FRAMES
  float*     G     = (float*)(ws + OFF_G);                 // reuses C1

  // --- tables ---
  k_hann  <<<8,     256, 0, stream>>>(hann);
  k_b2    <<<16384, 256, 0, stream>>>(b2);
  k_icos  <<<8448,  256, 0, stream>>>(icos);

  // --- STFT: frames -> WMMA GEMM -> magnitude ---
  k_frames<<<32768, 256, 0, stream>>>(audio, hann, F);
  {
    int M = NB * NFR, N = WIN, K = WIN;                    // 4096 x 2048 x 2048
    int waves = (M >> 4) * (N >> 4);
    k_gemm<<<waves / 8, 256, 0, stream>>>(F, b2, C1, M, N, K);
  }
  k_spec_mag<<<16384, 256, 0, stream>>>(C1, spec);

  // --- projections + dilated blocks ---
  k_proj_in    <<<512,  256, 0, stream>>>(spec, w_in, b_in, h);
  k_blocks     <<<NB,   256, 0, stream>>>(h, bw, bb, acc);
  k_proj_sparse<<<4096, 256, 0, stream>>>(acc, psw, psb, out_sparse);
  k_proj_dense <<<512,  256, 0, stream>>>(out_sparse, pdw, pdb, xv);

  // --- resonance synthesis: mags -> WMMA iDFT GEMM -> overlap-add ---
  k_mags<<<(CHN * KPAD + 255) / 256, 256, 0, stream>>>(reso, mags);
  {
    int M = CHN * NFR, N = WIN, K = KPAD;                  // 4096 x 2048 x 1056
    int waves = (M >> 4) * (N >> 4);
    k_gemm<<<waves / 8, 256, 0, stream>>>(mags, icos, G, M, N, K);
  }
  k_ola<<<4096, 256, 0, stream>>>(G, hann, res);

  // --- 128-tap sparse convolution + channel sum ---
  k_conv<<<NB * 128, 256, 0, stream>>>(xv, res, out_y);
}
